// FusedAttention_46926812677005
// MI455X (gfx1250) — matrix operand order repaired and checked by reading, not compiled, not run on hardware
//
#include <hip/hip_runtime.h>

// ---------------- problem constants ----------------
#define BATCH    2
#define SEQ      2048
#define DIM      4096
#define NH       32
#define NKV      8
#define HD       128
#define QD       4096           // NH*HD
#define KD       1024           // NKV*HD
#define QKVD     6144           // QD + 2*KD
#define MROWS    4096           // BATCH*SEQ

typedef __attribute__((ext_vector_type(16))) __bf16 v16bf;
typedef __attribute__((ext_vector_type(8)))  float  v8f;
typedef int v4i __attribute__((vector_size(16)));   // matches async-LDS builtin param

#define AS1 __attribute__((address_space(1)))
#define AS3 __attribute__((address_space(3)))

union FragU { v16bf v; struct { uint4 lo; uint4 hi; } u; };

// fp32 -> bf16 round-to-nearest-even
__device__ __forceinline__ __bf16 f2bf(float f) {
    unsigned int u = __builtin_bit_cast(unsigned int, f);
    unsigned int lsb = (u >> 16) & 1u;
    u += 0x7fffu + lsb;
    unsigned short h = (unsigned short)(u >> 16);
    return __builtin_bit_cast(__bf16, h);
}
__device__ __forceinline__ float bf2f(__bf16 b) {
    unsigned short h = __builtin_bit_cast(unsigned short, b);
    unsigned int u = ((unsigned int)h) << 16;
    return __builtin_bit_cast(float, u);
}

// 16-byte global -> LDS copy, async (ASYNCcnt) when the gfx1250 builtin exists.
__device__ __forceinline__ void cp16_g2l(const __bf16* g, __bf16* l) {
#if __has_builtin(__builtin_amdgcn_global_load_async_to_lds_b128)
    __builtin_amdgcn_global_load_async_to_lds_b128(
        (AS1 v4i*)g, (AS3 v4i*)l, 0, 0);
#else
    *(uint4*)l = *(const uint4*)g;
#endif
}
__device__ __forceinline__ void wait_async_lds() {
#if __has_builtin(__builtin_amdgcn_s_wait_asynccnt)
    __builtin_amdgcn_s_wait_asynccnt(0);
#else
    asm volatile("s_wait_asynccnt 0" ::: "memory");
#endif
}

// Load a 16x32 bf16 fragment from a row-major tile (row stride ld elements).
// Matches CDNA5 16-bit A/B layout: lane&15 = row (A) / col (B), lane>>4 picks
// K halves: VGPR0-3 <- K 0..7 (lanes 0-15) / 8..15 (lanes 16-31), VGPR4-7 <- +16.
__device__ __forceinline__ v16bf load_frag(const __bf16* __restrict__ p, int ld, int lane) {
    const __bf16* q = p + (long)(lane & 15) * ld + ((lane >> 4) << 3);
    FragU f;
    f.u.lo = *(const uint4*)q;
    f.u.hi = *(const uint4*)(q + 16);
    return f.v;
}

// ---------------- cast kernel ----------------
__global__ __launch_bounds__(256) void cast_f32_bf16(const float* __restrict__ src,
                                                     __bf16* __restrict__ dst, long n) {
    long i = (long)blockIdx.x * 256 + threadIdx.x;
    if (i < n) dst[i] = f2bf(src[i]);
}

// ---------------- bf16 WMMA GEMM with async-LDS double buffering: C = A * B^T ----
// A: [M,K] row-major bf16 (lda), Bw: [N,K] row-major bf16 (ldb).
// 128x128 tile per block, K-slab 64, 256 threads = 8 waves in 4x2; wave does 32x64.
// LDS: 2 x (128x64 A + 128x64 B) bf16 = 64 KB of the 320 KB WGP LDS.
template <bool OUT_F32>
__global__ __launch_bounds__(256)
void gemm_bf16_wmma(const __bf16* __restrict__ A, const __bf16* __restrict__ Bw,
                    void* __restrict__ Cout, int K, int lda, int ldb, int ldc) {
    __shared__ __align__(16) __bf16 shA[2][128 * 64];
    __shared__ __align__(16) __bf16 shB[2][128 * 64];

    const int tid  = threadIdx.x;
    const int lane = tid & 31;
    const int wave = tid >> 5;
    const int wr = wave & 3;           // 0..3 -> 32-row strip
    const int wc = wave >> 2;          // 0..1 -> 64-col strip
    const long blockM = (long)blockIdx.y * 128;
    const long blockN = (long)blockIdx.x * 128;
    const int rowW = wr * 32;
    const int colW = wc * 64;

    v8f acc[2][4];
    v8f zero = {};
#pragma unroll
    for (int i = 0; i < 2; ++i)
#pragma unroll
        for (int j = 0; j < 4; ++j) acc[i][j] = zero;

    // Stage one 128x64 slab of A and B into LDS buffer `buf` (16 B per copy,
    // 1024 chunks per matrix, 4 per thread each).
    auto stage = [&](int k0, int buf) {
#pragma unroll
        for (int c = 0; c < 4; ++c) {
            int id  = tid + c * 256;          // 0..1023
            int row = id >> 3;                // 0..127
            int cc  = (id & 7) << 3;          // 0,8,...,56 (bf16 elems)
            cp16_g2l(A  + (blockM + row) * (long)lda + k0 + cc, &shA[buf][row * 64 + cc]);
            cp16_g2l(Bw + (blockN + row) * (long)ldb + k0 + cc, &shB[buf][row * 64 + cc]);
        }
    };

    stage(0, 0);
    wait_async_lds();
    __syncthreads();

    const int nIter = K >> 6;
    for (int it = 0; it < nIter; ++it) {
        const int buf = it & 1;
        if (it + 1 < nIter) stage((it + 1) << 6, buf ^ 1);   // prefetch next slab

        const __bf16* tA = &shA[buf][0];
        const __bf16* tB = &shB[buf][0];
#pragma unroll
        for (int kk = 0; kk < 64; kk += 32) {
            v16bf a[2];
#pragma unroll
            for (int i = 0; i < 2; ++i)
                a[i] = load_frag(tA + (rowW + 16 * i) * 64 + kk, 64, lane);
            v16bf bfr[4];
#pragma unroll
            for (int j = 0; j < 4; ++j)
                bfr[j] = load_frag(tB + (colW + 16 * j) * 64 + kk, 64, lane);
#pragma unroll
            for (int i = 0; i < 2; ++i)
#pragma unroll
                for (int j = 0; j < 4; ++j)
                    acc[i][j] = __builtin_amdgcn_wmma_f32_16x16x32_bf16(
                        false, a[i], false, bfr[j], (short)0, acc[i][j], false, false);
        }
        wait_async_lds();
        __syncthreads();
    }

    // C layout: VGPR v, lanes 0-15 -> row v, lanes 16-31 -> row v+8
    const int cn = lane & 15;
    const int rh = (lane >> 4) << 3;
#pragma unroll
    for (int i = 0; i < 2; ++i)
#pragma unroll
        for (int j = 0; j < 4; ++j) {
            long r0 = blockM + rowW + 16 * i + rh;
            long c  = blockN + colW + 16 * j + cn;
#pragma unroll
            for (int v = 0; v < 8; ++v) {
                if (OUT_F32)
                    ((float*)Cout)[(r0 + v) * (long)ldc + c] = acc[i][j][v];
                else
                    ((__bf16*)Cout)[(r0 + v) * (long)ldc + c] = f2bf(acc[i][j][v]);
            }
        }
}

// ---------------- RoPE in place on bf16 qkv (q heads 0..31, k heads 32..39) ----------------
__global__ __launch_bounds__(256) void rope_kernel(__bf16* __restrict__ qkv,
                                                   const float* __restrict__ cosT,
                                                   const float* __restrict__ sinT) {
    long idx = (long)blockIdx.x * 256 + threadIdx.x;   // [B*T * 40 heads * 64 pairs)
    const long total = (long)BATCH * SEQ * 40 * 64;
    if (idx >= total) return;
    int p = idx & 63;
    long t2 = idx >> 6;
    int hh = (int)(t2 % 40);
    long bt = t2 / 40;                  // 0..MROWS-1
    int t = (int)(bt & (SEQ - 1));
    float c = cosT[t * 64 + p];
    float s = sinT[t * 64 + p];
    long col = (hh < 32) ? (long)hh * HD : (long)QD + (long)(hh - 32) * HD;
    __bf16* ptr = qkv + bt * QKVD + col + 2 * p;
    float tr = bf2f(ptr[0]);
    float ti = bf2f(ptr[1]);
    ptr[0] = f2bf(tr * c - ti * s);
    ptr[1] = f2bf(tr * s + ti * c);
}

// ---------------- V transpose: vt[b][hk][d][t] = qkv[b*T+t][5120 + hk*128 + d] ----------------
__global__ __launch_bounds__(256) void vtrans_kernel(const __bf16* __restrict__ qkv,
                                                     __bf16* __restrict__ vt) {
    long idx = (long)blockIdx.x * 256 + threadIdx.x;   // B*NKV*HD*T
    const long total = (long)BATCH * NKV * HD * SEQ;
    if (idx >= total) return;
    int t = (int)(idx & (SEQ - 1));
    long r = idx >> 11;
    int d = (int)(r & (HD - 1));
    long r2 = r >> 7;
    int hk = (int)(r2 & (NKV - 1));
    int b = (int)(r2 >> 3);
    vt[idx] = qkv[((long)(b * SEQ + t)) * QKVD + (QD + KD) + hk * HD + d];
}

// ---------------- Flash attention: one block = (b, h, 64 q rows), 4 waves x 16 rows ----------------
__global__ __launch_bounds__(128)
void attn_kernel(const __bf16* __restrict__ qkv, const __bf16* __restrict__ vt,
                 __bf16* __restrict__ yb) {
    const int lane = threadIdx.x & 31;
    const int wave = threadIdx.x >> 5;
    const int q0   = blockIdx.x * 64;
    const int h    = blockIdx.y;
    const int b    = blockIdx.z;
    const int hk   = h >> 2;                 // GQA: 4 q-heads per kv-head
    const int qr0  = q0 + wave * 16;
    const long rowBase = (long)(b * SEQ + qr0);
    const float scale = 0.08838834764831845f;  // 1/sqrt(128)

    __shared__ __align__(16) __bf16 psh[4][16 * 64];

    // Q fragments: 16 rows x 128 hd as 4 chunks of K=32
    v16bf qf[4];
#pragma unroll
    for (int kc = 0; kc < 4; ++kc)
        qf[kc] = load_frag(qkv + rowBase * QKVD + (long)h * HD + kc * 32, QKVD, lane);

    v8f o[8];
    v8f zero = {};
#pragma unroll
    for (int nc = 0; nc < 8; ++nc) o[nc] = zero;
    float mrow[8], lrow[8];
#pragma unroll
    for (int i = 0; i < 8; ++i) { mrow[i] = -3.0e38f; lrow[i] = 0.f; }

    const int cn = lane & 15;
    const int rloc = (lane >> 4) << 3;
    const int kbEnd = q0 + 64;               // same trip count for all 4 waves

    for (int kb = 0; kb < kbEnd; kb += 64) {
        // ---- S = Q * K^T over a 64-key block (4 n-frags x 4 k-chunks) ----
        v8f s[4];
#pragma unroll
        for (int nf = 0; nf < 4; ++nf) s[nf] = zero;
#pragma unroll
        for (int nf = 0; nf < 4; ++nf) {
#pragma unroll
            for (int kc = 0; kc < 4; ++kc) {
                v16bf kf = load_frag(
                    qkv + (long)(b * SEQ + kb + nf * 16) * QKVD + QD + (long)hk * HD + kc * 32,
                    QKVD, lane);
                s[nf] = __builtin_amdgcn_wmma_f32_16x16x32_bf16(
                    false, qf[kc], false, kf, (short)0, s[nf], false, false);
            }
        }

        // ---- mask + online softmax (row i lives in VGPR i, lanes split by rloc) ----
#pragma unroll
        for (int i = 0; i < 8; ++i) {
            int r = qr0 + i + rloc;
            float mv = -3.0e38f;
#pragma unroll
            for (int nf = 0; nf < 4; ++nf) {
                int c = kb + nf * 16 + cn;
                float sv = s[nf][i] * scale;
                sv = (c <= r) ? sv : -3.0e38f;
                s[nf][i] = sv;
                mv = fmaxf(mv, sv);
            }
#pragma unroll
            for (int off = 1; off < 16; off <<= 1)
                mv = fmaxf(mv, __shfl_xor(mv, off, 32));
            float mnew = fmaxf(mrow[i], mv);
            float alpha = __expf(mrow[i] - mnew);
            float lsum = 0.f;
#pragma unroll
            for (int nf = 0; nf < 4; ++nf) {
                float pv = __expf(s[nf][i] - mnew);
                s[nf][i] = pv;
                lsum += pv;
            }
#pragma unroll
            for (int off = 1; off < 16; off <<= 1)
                lsum += __shfl_xor(lsum, off, 32);
            lrow[i] = lrow[i] * alpha + lsum;
            mrow[i] = mnew;
#pragma unroll
            for (int nc = 0; nc < 8; ++nc) o[nc][i] *= alpha;
        }

        // ---- P (C-layout f32) -> LDS bf16 (row-major 16x64) -> A fragments ----
#pragma unroll
        for (int nf = 0; nf < 4; ++nf)
#pragma unroll
            for (int i = 0; i < 8; ++i)
                psh[wave][(i + rloc) * 64 + nf * 16 + cn] = f2bf(s[nf][i]);
        __syncthreads();

        v16bf pf[2];
#pragma unroll
        for (int kc = 0; kc < 2; ++kc)
            pf[kc] = load_frag(&psh[wave][kc * 32], 64, lane);

        // ---- O += P * V (V pre-transposed: vt[b][hk][d][t]) ----
#pragma unroll
        for (int nc = 0; nc < 8; ++nc) {
#pragma unroll
            for (int kc = 0; kc < 2; ++kc) {
                v16bf vf = load_frag(
                    vt + ((long)(b * NKV + hk) * HD + nc * 16) * SEQ + kb + kc * 32,
                    SEQ, lane);
                o[nc] = __builtin_amdgcn_wmma_f32_16x16x32_bf16(
                    false, pf[kc], false, vf, (short)0, o[nc], false, false);
            }
        }
        __syncthreads();
    }

    // ---- epilogue: O /= l, write bf16 to y[b*T+q][h*128 + d] ----
#pragma unroll
    for (int i = 0; i < 8; ++i) {
        float inv = 1.f / fmaxf(lrow[i], 1e-30f);
#pragma unroll
        for (int nc = 0; nc < 8; ++nc) {
            long row = rowBase + i + rloc;
            long col = (long)h * HD + nc * 16 + cn;
            yb[row * (long)DIM + col] = f2bf(o[nc][i] * inv);
        }
    }
}

// ---------------- host launch ----------------
extern "C" void kernel_launch(void* const* d_in, const int* in_sizes, int n_in,
                              void* d_out, int out_size, void* d_ws, size_t ws_size,
                              hipStream_t stream) {
    const float* x    = (const float*)d_in[0];
    // d_in[1] = freqs_cis (unused by reference math)
    const float* wqkv = (const float*)d_in[2];
    const float* wo   = (const float*)d_in[3];
    const float* cosT = (const float*)d_in[4];
    const float* sinT = (const float*)d_in[5];
    float* out = (float*)d_out;

    char* ws = (char*)d_ws;
    const size_t SZ_XB    = (size_t)MROWS * DIM * 2;     // 33.5 MB
    const size_t SZ_WQKVB = (size_t)QKVD * DIM * 2;      // 50.3 MB
    const size_t SZ_WOB   = (size_t)DIM * DIM * 2;       // 33.5 MB
    const size_t SZ_QKVB  = (size_t)MROWS * QKVD * 2;    // 50.3 MB
    const size_t SZ_VT    = (size_t)BATCH * NKV * HD * SEQ * 2; // 8.4 MB
    __bf16* xb    = (__bf16*)(ws);
    __bf16* wqkvb = (__bf16*)(ws + SZ_XB);
    __bf16* wob   = (__bf16*)(ws + SZ_XB + SZ_WQKVB);
    __bf16* qkvb  = (__bf16*)(ws + SZ_XB + SZ_WQKVB + SZ_WOB);
    __bf16* vtb   = (__bf16*)(ws + SZ_XB + SZ_WQKVB + SZ_WOB + SZ_QKVB);
    __bf16* yb    = (__bf16*)(ws + SZ_XB + SZ_WQKVB + SZ_WOB + SZ_QKVB + SZ_VT);

    // 1. casts to bf16
    {
        long n = (long)MROWS * DIM;
        cast_f32_bf16<<<(n + 255) / 256, 256, 0, stream>>>(x, xb, n);
        n = (long)QKVD * DIM;
        cast_f32_bf16<<<(n + 255) / 256, 256, 0, stream>>>(wqkv, wqkvb, n);
        n = (long)DIM * DIM;
        cast_f32_bf16<<<(n + 255) / 256, 256, 0, stream>>>(wo, wob, n);
    }

    // 2. qkv = x * wqkv^T   (M=4096, N=6144, K=4096)
    gemm_bf16_wmma<false><<<dim3(QKVD / 128, MROWS / 128), 256, 0, stream>>>(
        xb, wqkvb, qkvb, DIM, DIM, DIM, QKVD);

    // 3. RoPE in place on q/k heads
    {
        long n = (long)BATCH * SEQ * 40 * 64;
        rope_kernel<<<(n + 255) / 256, 256, 0, stream>>>(qkvb, cosT, sinT);
    }

    // 4. transpose V for the PV matmul
    {
        long n = (long)BATCH * NKV * HD * SEQ;
        vtrans_kernel<<<(n + 255) / 256, 256, 0, stream>>>(qkvb, vtb);
    }

    // 5. flash attention -> yb
    attn_kernel<<<dim3(SEQ / 64, NH, BATCH), 128, 0, stream>>>(qkvb, vtb, yb);

    // 6. out = y * wo^T  (f32 output)
    gemm_bf16_wmma<true><<<dim3(DIM / 128, MROWS / 128), 256, 0, stream>>>(
        yb, wob, out, DIM, DIM, DIM, DIM);

    (void)in_sizes; (void)n_in; (void)out_size; (void)ws_size;
}
